// TokaInferAttention_9844065042738
// MI455X (gfx1250) — matrix operand order repaired and checked by reading, not compiled, not run on hardware
//
#include <hip/hip_runtime.h>
#include <hip/hip_bf16.h>
#include <math.h>

// ---------------------------------------------------------------------------
// TokaInferAttention for MI455X (gfx1250, wave32, WMMA + TDM)
//   B=16 T=64 H=32 Hk=8 (G=4) D=128 C=4096 S=4160
// Phase 1: RMSNorm + RoPE on q,k (+ bf16 cast of new v) -> workspace (bf16)
// Phase 2: flash attention, 1 block per (b, hk), 512 threads = 16 wave32s,
//          each wave owns one 16-row M-tile (keeps VGPRs < 256).
//          K/V f32 cache chunks are DMA'd HBM->LDS by the Tensor Data Mover
//          (double buffered, tensor_load_to_lds + s_wait_tensorcnt),
//          converted LDS->LDS to bf16, consumed by v_wmma_f32_16x16x32_bf16.
// ---------------------------------------------------------------------------

typedef __attribute__((ext_vector_type(16))) __bf16 v16bf;
typedef __attribute__((ext_vector_type(8)))  float  v8f;
typedef __attribute__((ext_vector_type(4)))  unsigned int v4u;
typedef __attribute__((ext_vector_type(8)))  int v8i;
typedef __attribute__((ext_vector_type(4)))  int v4i;

union BF16x16 { v16bf v; unsigned u[8]; };

#define DEV __device__ __forceinline__

static constexpr int Bc  = 16;
static constexpr int Tc  = 64;
static constexpr int Hc  = 32;
static constexpr int HKc = 8;
static constexpr int Dc  = 128;
static constexpr int Cc  = 4096;

DEV unsigned f2bf(float f) {                 // f32 -> bf16 bits, round-nearest-even
  unsigned u = __float_as_uint(f);
  return (u + 0x7FFFu + ((u >> 16) & 1u)) >> 16;
}
DEV unsigned pack2(float lo, float hi) { return f2bf(lo) | (f2bf(hi) << 16); }

DEV v8f vzero() {
  v8f z;
#pragma unroll
  for (int i = 0; i < 8; ++i) z[i] = 0.0f;
  return z;
}

// ---------------------------------------------------------------------------
// TDM: load a 64-row x 128-elem (f32) tile, row stride Hk*D elems, into LDS.
// D# layout per CDNA5 ISA ch.8 (group0 = 128b, group1 = 256b, groups 2/3 = 0).
// This toolchain exposes the 6-arg builtin (g0, g1, g2, g3, g4ext, cpol).
// ---------------------------------------------------------------------------
DEV void tdm_load_tile_f32(unsigned long long gaddr, unsigned lds_off) {
  v4u g0;
  g0[0] = 1u;                                          // count=1, user mode
  g0[1] = lds_off;                                     // lds_addr (bytes)
  g0[2] = (unsigned)(gaddr & 0xFFFFFFFFu);             // global_addr[31:0]
  g0[3] = (unsigned)((gaddr >> 32) & 0x1FFFFFFu)       // global_addr[56:32]
          | (2u << 30);                                // type=2 ("image")
  v8i g1;
  g1[0] = (int)(2u << 16);                             // data_size=2 (4 bytes)
  g1[1] = (int)(128u << 16);                           // tensor_dim0 = 128
  g1[2] = (int)(64u  << 16);                           // tensor_dim1 = 64
  g1[3] = (int)(128u << 16);                           // tile_dim0 = 128
  g1[4] = 64;                                          // tile_dim1 = 64
  g1[5] = (int)(HKc * Dc);                             // dim0 stride = 1024 elems
  g1[6] = 0;
  g1[7] = 0;
  v4i gz4; gz4[0] = 0; gz4[1] = 0; gz4[2] = 0; gz4[3] = 0;
  v8i gz8;
#pragma unroll
  for (int i = 0; i < 8; ++i) gz8[i] = 0;
  __builtin_amdgcn_tensor_load_to_lds(g0, g1, gz4, gz4, gz8, 0);
}

// ---------------------------------------------------------------------------
// Kernel 1: RMSNorm + RoPE (q rows then k rows; k rows also cast v -> bf16).
// One wave32 per 128-wide row; lane owns d = {lane, lane+32, lane+64, lane+96}
// which is exactly the RoPE pairing (i, i+64).
// ---------------------------------------------------------------------------
__global__ __launch_bounds__(256) void toka_prep_kernel(
    const float* __restrict__ q, const float* __restrict__ k,
    const float* __restrict__ v,
    const float* __restrict__ q_scale, const float* __restrict__ k_scale,
    const int* __restrict__ cache_len,
    unsigned short* __restrict__ qn, unsigned short* __restrict__ kn,
    unsigned short* __restrict__ vn)
{
  const int lane = threadIdx.x & 31;
  const int wid  = threadIdx.x >> 5;
  const int row  = blockIdx.x * 8 + wid;
  const int QROWS = Bc * Tc * Hc;          // 32768
  const int KROWS = Bc * Tc * HKc;         // 8192
  if (row >= QROWS + KROWS) return;

  const float* src;
  const float* scale;
  unsigned short* dst;
  int b, t;
  if (row < QROWS) {
    b = row / (Tc * Hc);
    t = (row / Hc) % Tc;
    src = q + (size_t)row * Dc;  scale = q_scale;  dst = qn + (size_t)row * Dc;
  } else {
    const int kr = row - QROWS;
    b = kr / (Tc * HKc);
    t = (kr / HKc) % Tc;
    src = k + (size_t)kr * Dc;  scale = k_scale;  dst = kn + (size_t)kr * Dc;
    const float* vs = v + (size_t)kr * Dc;
    unsigned short* vd = vn + (size_t)kr * Dc;
#pragma unroll
    for (int j = 0; j < 4; ++j)
      vd[lane + 32 * j] = (unsigned short)f2bf(vs[lane + 32 * j]);
  }

  const float x0 = src[lane], x1 = src[lane + 32],
              x2 = src[lane + 64], x3 = src[lane + 96];
  float ss = x0 * x0 + x1 * x1 + x2 * x2 + x3 * x3;
#pragma unroll
  for (int off = 16; off >= 1; off >>= 1) ss += __shfl_xor(ss, off, 32);
  const float rn = rsqrtf(ss * (1.0f / 128.0f) + 1e-6f);

  const float y0 = x0 * rn * scale[lane];
  const float y1 = x1 * rn * scale[lane + 32];
  const float y2 = x2 * rn * scale[lane + 64];
  const float y3 = x3 * rn * scale[lane + 96];

  const float pos = (float)(cache_len[b] + t);
  const float LOG2_BASE = 13.287712379549449f;        // log2(10000)
  const float f0 = exp2f(-((float)lane)        * (1.0f / 64.0f) * LOG2_BASE);
  const float f1 = exp2f(-((float)(lane + 32)) * (1.0f / 64.0f) * LOG2_BASE);
  const float a0 = pos * f0, a1 = pos * f1;
  const float c0 = __cosf(a0), s0 = __sinf(a0);
  const float c1 = __cosf(a1), s1 = __sinf(a1);

  dst[lane]      = (unsigned short)f2bf(y0 * c0 - y2 * s0);
  dst[lane + 64] = (unsigned short)f2bf(y0 * s0 + y2 * c0);
  dst[lane + 32] = (unsigned short)f2bf(y1 * c1 - y3 * s1);
  dst[lane + 96] = (unsigned short)f2bf(y1 * s1 + y3 * c1);
}

// ---------------------------------------------------------------------------
// Kernel 2: flash attention. grid = B*Hk (128 blocks), 512 threads (16 waves).
// Wave w owns query rows [w*16, w*16+16) of the 256 (g,t) rows of this kv head.
// Cache chunks: TDM-prefetched (double-buffered f32) -> bf16 convert -> WMMA.
// ---------------------------------------------------------------------------
__global__ __launch_bounds__(512) void toka_attn_kernel(
    const float* __restrict__ k_cache, const float* __restrict__ v_cache,
    const unsigned short* __restrict__ qn, const unsigned short* __restrict__ kn,
    const unsigned short* __restrict__ vn, const int* __restrict__ cache_len,
    float* __restrict__ out)
{
  __shared__ __align__(16) float          Kf32[2][64 * 128]; // ping-pong 64 KB
  __shared__ __align__(16) float          Vf32[2][64 * 128]; // ping-pong 64 KB
  __shared__ __align__(16) unsigned short Klds[64 * 128];    // [s][d]  16 KB
  __shared__ __align__(16) unsigned short Vlds[128 * 64];    // [d][s]  16 KB
  __shared__ __align__(16) unsigned short Plds[16][16 * 64]; // per-wave 32 KB

  const int tid  = threadIdx.x;
  const int lane = tid & 31;
  const int wid  = tid >> 5;       // 0..15
  const int r    = lane & 15;      // column-lane within WMMA tile
  const int hi   = lane >> 4;      // lane half
  const int b    = blockIdx.x >> 3;
  const int hk   = blockIdx.x & 7;

  int cvalid = cache_len[b];
  cvalid = cvalid < 0 ? 0 : (cvalid > Cc ? Cc : cvalid);

  const int g  = wid >> 2;               // query head within group, 0..3
  const int h  = hk * 4 + g;
  const int tw = (wid & 3) * 16;         // t-base of this wave within head g

  const float* kbase = k_cache + ((size_t)b * Cc * HKc + hk) * Dc;
  const float* vbase = v_cache + ((size_t)b * Cc * HKc + hk) * Dc;

  // ---- preload Q fragments (16-bit A layout, ISA 7.12.2) ----
  BF16x16 qf[4];
  {
    const int t = tw + r;
    const unsigned short* qrow = qn + (((size_t)(b * Tc + t) * Hc + h) * Dc);
#pragma unroll
    for (int kk = 0; kk < 4; ++kk)
#pragma unroll
      for (int j = 0; j < 8; ++j) {
        const int d = kk * 32 + hi * 8 + (j < 4 ? 2 * j : 16 + 2 * (j - 4));
        qf[kk].u[j] = *(const unsigned*)(qrow + d);
      }
  }

  v8f   oacc[8];
  float mrow[8], lrow[8];
#pragma unroll
  for (int i = 0; i < 8; ++i) { mrow[i] = -3.0e38f; lrow[i] = 0.0f; }
#pragma unroll
  for (int nt = 0; nt < 8; ++nt) oacc[nt] = vzero();

  unsigned short* Pw = Plds[wid];
  const int ncc = (cvalid + 63) >> 6;        // # cache chunks actually needed

  // ---- TDM prologue: start DMA of chunk 0 into buffer 0 ----
  if (wid == 0 && ncc > 0) {
    tdm_load_tile_f32((unsigned long long)(size_t)kbase,
                      (unsigned)(size_t)&Kf32[0][0]);
    tdm_load_tile_f32((unsigned long long)(size_t)vbase,
                      (unsigned)(size_t)&Vf32[0][0]);
  }

  for (int it = 0; it <= ncc; ++it) {
    const bool isnew = (it == ncc);          // final chunk = the T new keys
    const int  cs    = isnew ? Cc : it * 64;
    const int  limit = isnew ? 64 : (cvalid - cs < 64 ? cvalid - cs : 64);
    const int  bufsel = it & 1;

    if (!isnew && wid == 0) {
      if (it + 1 < ncc) {                    // DMA next cache chunk, other buffer
        const size_t roff = (size_t)(it + 1) * 64 * HKc * Dc;
        tdm_load_tile_f32((unsigned long long)(size_t)(kbase + roff),
                          (unsigned)(size_t)&Kf32[bufsel ^ 1][0]);
        tdm_load_tile_f32((unsigned long long)(size_t)(vbase + roff),
                          (unsigned)(size_t)&Vf32[bufsel ^ 1][0]);
        __builtin_amdgcn_s_wait_tensorcnt(2);   // chunk `it` landed
      } else {
        __builtin_amdgcn_s_wait_tensorcnt(0);
      }
    }
    __syncthreads();   // TDM data visible to all waves; prev compute finished

    // ---- convert f32 tile -> bf16 K [s][d] and V^T [d][s] ----
    {
      const int sr = tid >> 3;               // key row in chunk, 0..63
      const int dq = (tid & 7) * 16;         // 16-wide d slice
      if (!isnew) {
        const float* kf = &Kf32[bufsel][sr * 128];
        const float* vf = &Vf32[bufsel][sr * 128];
#pragma unroll
        for (int j = 0; j < 8; ++j) {
          const int d = dq + 2 * j;
          ((unsigned*)Klds)[(sr * 128 + d) >> 1] = pack2(kf[d], kf[d + 1]);
          Vlds[(d    ) * 64 + sr] = (unsigned short)f2bf(vf[d]);
          Vlds[(d + 1) * 64 + sr] = (unsigned short)f2bf(vf[d + 1]);
        }
      } else {                               // new keys: already bf16 in ws
        const unsigned short* kr_ = kn + (((size_t)(b * Tc + sr)) * HKc + hk) * Dc;
        const unsigned short* vr_ = vn + (((size_t)(b * Tc + sr)) * HKc + hk) * Dc;
#pragma unroll
        for (int j = 0; j < 8; ++j) {
          const int d = dq + 2 * j;
          ((unsigned*)Klds)[(sr * 128 + d) >> 1] = *(const unsigned*)(kr_ + d);
          Vlds[(d    ) * 64 + sr] = vr_[d];
          Vlds[(d + 1) * 64 + sr] = vr_[d + 1];
        }
      }
    }
    __syncthreads();

    // ---- scores S = Q * K^T : per-wave 16x64, 16 WMMAs ----
    // Load all 4 K-dim fragments of a column tile first (8 ds_load_b128 in
    // flight), then run the 4 chained WMMAs -> partial dscnt waits, not 0.
    v8f sacc[4];
#pragma unroll
    for (int nt = 0; nt < 4; ++nt) {
      BF16x16 kb[4];                          // B fragments (16-bit B layout)
#pragma unroll
      for (int kk = 0; kk < 4; ++kk)
#pragma unroll
        for (int j = 0; j < 8; ++j) {
          const int d = kk * 32 + hi * 8 + (j < 4 ? 2 * j : 16 + 2 * (j - 4));
          kb[kk].u[j] = *(const unsigned*)(Klds + (nt * 16 + r) * 128 + d);
        }
      v8f c = vzero();
#pragma unroll
      for (int kk = 0; kk < 4; ++kk)
        c = __builtin_amdgcn_wmma_f32_16x16x32_bf16(
                false, qf[kk].v, false, kb[kk].v, (short)0, c, false, false);
      sacc[nt] = c;
    }

    // ---- mask + scale + online softmax (rows live at (hi*8+vi) per C layout) ----
    {
      float mnew[8];
#pragma unroll
      for (int vi = 0; vi < 8; ++vi) mnew[vi] = mrow[vi];
#pragma unroll
      for (int nt = 0; nt < 4; ++nt) {
        const int col = nt * 16 + r;
#pragma unroll
        for (int vi = 0; vi < 8; ++vi) {
          float val = sacc[nt][vi] * 0.08838834764831845f;   // D^-0.5
          const int trow = tw + hi * 8 + vi;
          const bool valid = isnew ? (col <= trow) : (col < limit);
          val = valid ? val : -3.0e38f;
          sacc[nt][vi] = val;
          mnew[vi] = fmaxf(mnew[vi], val);
        }
      }
#pragma unroll
      for (int off = 8; off >= 1; off >>= 1)
#pragma unroll
        for (int vi = 0; vi < 8; ++vi)
          mnew[vi] = fmaxf(mnew[vi], __shfl_xor(mnew[vi], off, 32));

      float alpha[8], lsum[8];
#pragma unroll
      for (int vi = 0; vi < 8; ++vi) {
        alpha[vi] = __expf(mrow[vi] - mnew[vi]);
        lsum[vi]  = 0.0f;
      }
#pragma unroll
      for (int nt = 0; nt < 4; ++nt)
#pragma unroll
        for (int vi = 0; vi < 8; ++vi) {
          const float p = __expf(sacc[nt][vi] - mnew[vi]);
          lsum[vi] += p;
          Pw[(hi * 8 + vi) * 64 + nt * 16 + r] = (unsigned short)f2bf(p);
        }
#pragma unroll
      for (int off = 8; off >= 1; off >>= 1)
#pragma unroll
        for (int vi = 0; vi < 8; ++vi)
          lsum[vi] += __shfl_xor(lsum[vi], off, 32);
#pragma unroll
      for (int vi = 0; vi < 8; ++vi) {
        lrow[vi] = lrow[vi] * alpha[vi] + lsum[vi];
        mrow[vi] = mnew[vi];
      }
#pragma unroll
      for (int nt = 0; nt < 8; ++nt)
#pragma unroll
        for (int vi = 0; vi < 8; ++vi)
          oacc[nt][vi] *= alpha[vi];
    }

    // ---- O += P * V : 16 WMMAs (P via LDS round-trip = C->A layout fix) ----
    // V fragments loaded two tiles at a time to keep 4 ds_load_b128 in flight.
#pragma unroll
    for (int kk = 0; kk < 2; ++kk) {
      BF16x16 pf;
#pragma unroll
      for (int j = 0; j < 8; ++j) {
        const int s = kk * 32 + hi * 8 + (j < 4 ? 2 * j : 16 + 2 * (j - 4));
        pf.u[j] = *(const unsigned*)(Pw + r * 64 + s);
      }
#pragma unroll
      for (int np = 0; np < 4; ++np) {
        BF16x16 vb[2];
#pragma unroll
        for (int q2 = 0; q2 < 2; ++q2)
#pragma unroll
          for (int j = 0; j < 8; ++j) {
            const int s = kk * 32 + hi * 8 + (j < 4 ? 2 * j : 16 + 2 * (j - 4));
            vb[q2].u[j] =
                *(const unsigned*)(Vlds + ((np * 2 + q2) * 16 + r) * 64 + s);
          }
#pragma unroll
        for (int q2 = 0; q2 < 2; ++q2)
          oacc[np * 2 + q2] = __builtin_amdgcn_wmma_f32_16x16x32_bf16(
              false, pf.v, false, vb[q2].v, (short)0, oacc[np * 2 + q2],
              false, false);
      }
    }
  }

  // ---- epilogue: normalize, scatter to (B*T, H*D) f32 ----
#pragma unroll
  for (int vi = 0; vi < 8; ++vi) {
    const int rowm = wid * 16 + hi * 8 + vi;            // g*64 + t
    const int gg = rowm >> 6, t = rowm & 63;
    const float inv = 1.0f / lrow[vi];
    const size_t base =
        (size_t)(b * Tc + t) * (Hc * Dc) + (size_t)(hk * 4 + gg) * Dc;
#pragma unroll
    for (int nt = 0; nt < 8; ++nt)
      out[base + nt * 16 + r] = oacc[nt][vi] * inv;
  }
}

// ---------------------------------------------------------------------------
extern "C" void kernel_launch(void* const* d_in, const int* in_sizes, int n_in,
                              void* d_out, int out_size, void* d_ws, size_t ws_size,
                              hipStream_t stream) {
  const float* q        = (const float*)d_in[0];
  const float* k        = (const float*)d_in[1];
  const float* v        = (const float*)d_in[2];
  const float* k_cache  = (const float*)d_in[3];
  const float* v_cache  = (const float*)d_in[4];
  const float* q_scale  = (const float*)d_in[5];
  const float* k_scale  = (const float*)d_in[6];
  const int*   cache_ln = (const int*)d_in[7];

  unsigned short* qn = (unsigned short*)d_ws;                  // 8.39 MB bf16
  unsigned short* kn = qn + (size_t)Bc * Tc * Hc * Dc;         // 2.10 MB bf16
  unsigned short* vn = kn + (size_t)Bc * Tc * HKc * Dc;        // 2.10 MB bf16

  const int nrows = Bc * Tc * Hc + Bc * Tc * HKc;              // 40960
  toka_prep_kernel<<<nrows / 8, 256, 0, stream>>>(
      q, k, v, q_scale, k_scale, cache_ln, qn, kn, vn);
  toka_attn_kernel<<<Bc * HKc, 512, 0, stream>>>(
      k_cache, v_cache, qn, kn, vn, cache_ln, (float*)d_out);
}
